// GroupedQueryAttention_79250736546372
// MI455X (gfx1250) — matrix operand order repaired and checked by reading, not compiled, not run on hardware
//
#include <hip/hip_runtime.h>
#include <hip/hip_bf16.h>
#include <cmath>

// ---------------------------------------------------------------------------
// GQA attention block for gfx1250 (MI455X), f16 WMMA with f32 accumulation.
//   B=2, T=2048, C=4096, NH=32, NKV=8, HD=128, GROUPS=4, theta=10000
// ---------------------------------------------------------------------------

typedef __attribute__((ext_vector_type(16))) _Float16 v16h;
typedef __attribute__((ext_vector_type(8)))  float    v8f;
typedef __attribute__((ext_vector_type(4)))  float    f4v;
typedef __attribute__((ext_vector_type(4)))  _Float16 h4v;
typedef __attribute__((ext_vector_type(2)))  _Float16 h2v;

#define B_      2
#define T_      2048
#define C_      4096
#define NH_     32
#define NKV_    8
#define HD_     128
#define GROUPS_ 4
#define BT_     (B_ * T_)          // 4096 rows of x

__device__ __forceinline__ v8f wmma_f16(v16h a, v16h b, v8f c) {
  return __builtin_amdgcn_wmma_f32_16x16x32_f16(false, a, false, b, (short)0, c,
                                                false, false);
}

// ---- WMMA fragment helpers (contiguous k = 16h + e, same order for A and B) ------
// A (16x32 f16): lane holds row m=lane&15; half e -> K = e + (lane&16)
__device__ __forceinline__ v16h load_a_frag(const _Float16* row, int lane) {
  v16h a;
  const int off = lane & 16;
#pragma unroll
  for (int e = 0; e < 16; ++e) a[e] = row[e + off];
  return a;
}
// B (32x16 f16) read from a K-contiguous (transposed) buffer:
// lane holds col n=lane&15; half e -> K = e + (lane&16)
__device__ __forceinline__ v16h load_b_frag(const _Float16* col, int lane) {
  v16h b;
  const int off = lane & 16;
#pragma unroll
  for (int e = 0; e < 16; ++e) b[e] = col[off + e];
  return b;
}

__device__ __forceinline__ float rowmax16(float x) {
#pragma unroll
  for (int i = 1; i < 16; i <<= 1) x = fmaxf(x, __shfl_xor(x, i, 16));
  return x;
}
__device__ __forceinline__ float rowsum16(float x) {
#pragma unroll
  for (int i = 1; i < 16; i <<= 1) x += __shfl_xor(x, i, 16);
  return x;
}

// ---------------------------------------------------------------------------
// Kernel 1: QKV projection GEMM.  out[b,h,t,d] (f16) = x[bt,:] @ w[:,h*HD+d]
// Block: 128 threads (4 waves), 128x64 output tile; each wave 32x64 (8 WMMA/step).
// ---------------------------------------------------------------------------
__global__ __launch_bounds__(128) void qkv_gemm_kernel(
    const float* __restrict__ x, const float* __restrict__ w,
    _Float16* __restrict__ out, int Ncols, int H) {
  __shared__ __align__(16) _Float16 lds_a[128 * 32];  // [row][k]
  __shared__ __align__(16) _Float16 lds_bt[64 * 32];  // [col][k] (transposed)

  const int tid  = threadIdx.x;
  const int lane = tid & 31;
  const int wave = tid >> 5;
  const int mBase = blockIdx.x * 128;
  const int nBase = blockIdx.y * 64;

  v8f acc[2][4] = {};

  for (int k0 = 0; k0 < C_; k0 += 32) {
    // ---- stage A: 128x32 f32 -> f16 via b128 loads, then packed cvt + b64 DS
    f4v ta[8];
#pragma unroll
    for (int s = 0; s < 8; ++s) {
      const int i = tid + 128 * s;                 // 1024 float4 groups
      const int r = i >> 3, c4 = (i & 7) * 4;
      ta[s] = *(const f4v*)&x[(size_t)(mBase + r) * C_ + k0 + c4];
    }
#pragma unroll
    for (int s = 0; s < 8; ++s) {
      const int i = tid + 128 * s;
      const int r = i >> 3, c4 = (i & 7) * 4;
      *(h4v*)&lds_a[r * 32 + c4] = __builtin_convertvector(ta[s], h4v);
    }
    // ---- stage B transposed: each thread owns 2(k) x 4(n) sub-blocks ----
    f4v tb[2][2];
#pragma unroll
    for (int s = 0; s < 2; ++s) {
      const int sb = tid + 128 * s;                // 256 sub-blocks
      const int r = (sb >> 4) * 2, c4 = (sb & 15) * 4;
      tb[s][0] = *(const f4v*)&w[(size_t)(k0 + r) * Ncols + nBase + c4];
      tb[s][1] = *(const f4v*)&w[(size_t)(k0 + r + 1) * Ncols + nBase + c4];
    }
#pragma unroll
    for (int s = 0; s < 2; ++s) {
      const int sb = tid + 128 * s;
      const int r = (sb >> 4) * 2, c4 = (sb & 15) * 4;
      const h4v h0 = __builtin_convertvector(tb[s][0], h4v);
      const h4v h1 = __builtin_convertvector(tb[s][1], h4v);
#pragma unroll
      for (int j = 0; j < 4; ++j) {
        h2v p; p.x = h0[j]; p.y = h1[j];           // k-pair -> one b32 DS store
        *(h2v*)&lds_bt[(c4 + j) * 32 + r] = p;
      }
    }
    if (k0 + 32 < C_) {  // global_prefetch_b8 for next k-slab
      __builtin_prefetch(&x[(size_t)(mBase + tid) * C_ + k0 + 32], 0, 0);
      __builtin_prefetch(&w[(size_t)(k0 + 32 + (tid & 31)) * Ncols + nBase], 0, 0);
    }
    __syncthreads();

    const v16h a0 = load_a_frag(&lds_a[(wave * 32 + (lane & 15)) * 32], lane);
    const v16h a1 = load_a_frag(&lds_a[(wave * 32 + 16 + (lane & 15)) * 32], lane);
#pragma unroll
    for (int nt = 0; nt < 4; ++nt) {
      const v16h b = load_b_frag(&lds_bt[(nt * 16 + (lane & 15)) * 32], lane);
      acc[0][nt] = wmma_f16(a0, b, acc[0][nt]);
      acc[1][nt] = wmma_f16(a1, b, acc[1][nt]);
    }
    __syncthreads();
  }

  // Scatter f16 into [B, H, T, HD]
  const int hi = (lane & 16) ? 8 : 0;
  const int n0 = lane & 15;
#pragma unroll
  for (int mt = 0; mt < 2; ++mt)
#pragma unroll
    for (int nt = 0; nt < 4; ++nt)
#pragma unroll
      for (int r = 0; r < 8; ++r) {
        const int m   = mBase + wave * 32 + mt * 16 + hi + r;
        const int col = nBase + nt * 16 + n0;
        const int bb = m / T_, t = m % T_;
        const int h = col / HD_, d = col % HD_;
        out[(((size_t)bb * H + h) * T_ + t) * HD_ + d] = (_Float16)acc[mt][nt][r];
      }
}

// ---------------------------------------------------------------------------
// Kernel 2: in-place RoPE on [B,H,T,HD] f16. One thread per (b,h,t,j<64) pair.
// ---------------------------------------------------------------------------
__global__ void rope_kernel(_Float16* __restrict__ q, int H) {
  const size_t idx = (size_t)blockIdx.x * blockDim.x + threadIdx.x;
  const size_t total = (size_t)B_ * H * T_ * 64;
  if (idx >= total) return;
  const int j = (int)(idx & 63);
  const size_t rest = idx >> 6;
  const int t = (int)(rest % T_);
  const size_t bh = rest / T_;
  const float inv_freq = __powf(10000.0f, -(float)j * (1.0f / 64.0f));
  const float ang = (float)t * inv_freq;
  float s, c;
  __sincosf(ang, &s, &c);
  _Float16* p = q + (bh * T_ + t) * HD_ + j;
  const float lo = (float)p[0], hi = (float)p[64];
  p[0]  = (_Float16)(lo * c - hi * s);
  p[64] = (_Float16)(hi * c + lo * s);
}

// ---------------------------------------------------------------------------
// Kernel 3: causal flash attention (GQA).
// Block = 4 waves / 64 queries sharing one staged K/V tile (amortizes LDS fill
// 4x vs one wave). K staged via GLOBAL_LOAD_ASYNC_TO_LDS_B128 (ASYNCcnt); V
// staged transposed. Per wave per 32-key step: 8 WMMA (S=QK^T) + 8 WMMA (P@V).
// Waves whose causal horizon excludes a key block skip compute via a
// wave-uniform guard (barriers stay outside; EXEC all-ones for WMMA).
// ---------------------------------------------------------------------------
__global__ __launch_bounds__(128) void attn_kernel(
    const _Float16* __restrict__ q, const _Float16* __restrict__ k,
    const _Float16* __restrict__ v, _Float16* __restrict__ attn_out) {
  __shared__ __align__(16) _Float16 ldsK[32 * 128];     // [key][d] row-major
  __shared__ __align__(16) _Float16 ldsVt[128 * 32];    // [d][key] transposed
  __shared__ __align__(16) _Float16 ldsP[4][16 * 32];   // per-wave P shuffle

  const int tid  = threadIdx.x;
  const int lane = tid & 31;
  const int wave = tid >> 5;
  const int qb_block = blockIdx.x * 64;       // 64 queries per block
  const int q_base = qb_block + wave * 16;    // this wave's 16 queries
  const int h = blockIdx.y;
  const int b = blockIdx.z;
  const int kvh = h / GROUPS_;
  const float scale = 0.08838834764831845f;   // 1/sqrt(128)

  // Load Q fragments once (4 chunks of K=32 along HD)
  const _Float16* qrow =
      q + (((size_t)b * NH_ + h) * T_ + q_base + (lane & 15)) * HD_;
  const int aoff = lane & 16;
  v16h qa[4];
#pragma unroll
  for (int c = 0; c < 4; ++c)
#pragma unroll
    for (int e = 0; e < 16; ++e) qa[c][e] = qrow[c * 32 + e + aoff];

  v8f O[8] = {};
  float rmax[8], rsum[8];
#pragma unroll
  for (int r = 0; r < 8; ++r) { rmax[r] = -__builtin_inff(); rsum[r] = 0.0f; }

  const _Float16* Kb = k + ((size_t)b * NKV_ + kvh) * T_ * HD_;
  const _Float16* Vb = v + ((size_t)b * NKV_ + kvh) * T_ * HD_;
  const int hi8 = (lane & 16) ? 8 : 0;
  const int n0 = lane & 15;
  // Key blocks needed by the block's last wave: keys <= qb_block+63
  const int nsteps = (qb_block + 64 + 31) / 32;

  for (int kb = 0; kb < nsteps; ++kb) {
    const int j0 = kb * 32;
    // ---- Stage K rows (32x128 f16 = 8KB) via async LDS DMA (ASYNCcnt) ----
    {
      const _Float16* src = Kb + (size_t)j0 * HD_;
#pragma unroll
      for (int s = 0; s < 4; ++s) {
        const int idx = tid + 128 * s;  // 512 x 16B
        const unsigned ldsoff = (unsigned)(size_t)&ldsK[idx * 8];
        const _Float16* g = src + idx * 8;
        asm volatile("global_load_async_to_lds_b128 %0, %1, off"
                     :: "v"(ldsoff), "v"(g) : "memory");
      }
    }
    // ---- Stage V transposed: ldsVt[d*32+key], batched load-then-store ----
    {
      const h2v* src = (const h2v*)(Vb + (size_t)j0 * HD_);
#pragma unroll
      for (int g = 0; g < 2; ++g) {
        h2v t[8];
#pragma unroll
        for (int u = 0; u < 8; ++u) t[u] = src[tid + 128 * (g * 8 + u)];
#pragma unroll
        for (int u = 0; u < 8; ++u) {
          const int i = tid + 128 * (g * 8 + u);
          const int key = i >> 6, dp = i & 63;
          ldsVt[(2 * dp + 0) * 32 + key] = t[u].x;
          ldsVt[(2 * dp + 1) * 32 + key] = t[u].y;
        }
      }
    }
    asm volatile("s_wait_asynccnt 0x0" ::: "memory");
    __syncthreads();  // all waves' K (async) and V (ds) now visible

    // Wave-uniform causal guard: any keys in [j0, j0+31] visible to this wave?
    if (j0 <= q_base + 15) {
      // S[nt] = Q @ K^T for keys [j0+16nt, j0+16nt+16)
      v8f S[2] = {};
#pragma unroll
      for (int nt = 0; nt < 2; ++nt) {
        const _Float16* kr = &ldsK[(nt * 16 + n0) * 128];
#pragma unroll
        for (int c = 0; c < 4; ++c) {
          v16h bf;
          const int boff = lane & 16;
#pragma unroll
          for (int e = 0; e < 16; ++e) bf[e] = kr[c * 32 + boff + e];
          S[nt] = wmma_f16(qa[c], bf, S[nt]);
        }
      }

      // Scale + causal mask; online softmax per row; P -> LDS (C-layout)
#pragma unroll
      for (int r = 0; r < 8; ++r) {
        const int mrow = q_base + hi8 + r;
        float s0 = S[0][r] * scale, s1 = S[1][r] * scale;
        if (j0 + n0 > mrow)      s0 = -__builtin_inff();
        if (j0 + 16 + n0 > mrow) s1 = -__builtin_inff();
        float mx = rowmax16(fmaxf(s0, s1));
        const float newmax = fmaxf(rmax[r], mx);
        const float alpha = __expf(rmax[r] - newmax);
        const float p0 = __expf(s0 - newmax);
        const float p1 = __expf(s1 - newmax);
        rsum[r] = rsum[r] * alpha + rowsum16(p0 + p1);
        rmax[r] = newmax;
#pragma unroll
        for (int t = 0; t < 8; ++t) O[t][r] *= alpha;
        ldsP[wave][(hi8 + r) * 32 + n0]      = (_Float16)p0;
        ldsP[wave][(hi8 + r) * 32 + 16 + n0] = (_Float16)p1;
      }

      // P (A-layout) @ V
      const v16h pa = load_a_frag(&ldsP[wave][n0 * 32], lane);
#pragma unroll
      for (int ht = 0; ht < 8; ++ht) {
        const v16h bf = load_b_frag(&ldsVt[(ht * 16 + n0) * 32], lane);
        O[ht] = wmma_f16(pa, bf, O[ht]);
      }
    }
    __syncthreads();  // all waves done with ldsK/ldsVt before restage
  }

  // Normalize and store f16 into [B*T, NH*HD]
#pragma unroll
  for (int r = 0; r < 8; ++r) {
    const float inv = 1.0f / rsum[r];
    const int t = q_base + hi8 + r;
#pragma unroll
    for (int ht = 0; ht < 8; ++ht) {
      const int d = ht * 16 + n0;
      attn_out[((size_t)(b * T_ + t)) * (NH_ * HD_) + h * HD_ + d] =
          (_Float16)(O[ht][r] * inv);
    }
  }
}

// ---------------------------------------------------------------------------
// Kernel 4: output projection. out(f32)[bt, c] = attn(f16) @ wo(f32->f16)
// Same 128x64 tiling as kernel 1; A is already f16 (pure b128 copy to LDS).
// ---------------------------------------------------------------------------
__global__ __launch_bounds__(128) void out_gemm_kernel(
    const _Float16* __restrict__ A, const float* __restrict__ W,
    float* __restrict__ out) {
  __shared__ __align__(16) _Float16 lds_a[128 * 32];
  __shared__ __align__(16) _Float16 lds_bt[64 * 32];

  const int tid  = threadIdx.x;
  const int lane = tid & 31;
  const int wave = tid >> 5;
  const int mBase = blockIdx.x * 128;
  const int nBase = blockIdx.y * 64;

  v8f acc[2][4] = {};

  for (int k0 = 0; k0 < C_; k0 += 32) {
    // stage A (f16): 128x32 halves, b128 copies
    uint4 ta[4];
#pragma unroll
    for (int s = 0; s < 4; ++s) {
      const int i = tid + 128 * s;                 // 512 x 8-half groups
      const int r = i >> 2, g8 = (i & 3) * 8;
      ta[s] = *(const uint4*)&A[(size_t)(mBase + r) * C_ + k0 + g8];
    }
#pragma unroll
    for (int s = 0; s < 4; ++s) {
      const int i = tid + 128 * s;
      const int r = i >> 2, g8 = (i & 3) * 8;
      *(uint4*)&lds_a[r * 32 + g8] = ta[s];
    }
    // stage B transposed (f32 -> f16)
    f4v tb[2][2];
#pragma unroll
    for (int s = 0; s < 2; ++s) {
      const int sb = tid + 128 * s;
      const int r = (sb >> 4) * 2, c4 = (sb & 15) * 4;
      tb[s][0] = *(const f4v*)&W[(size_t)(k0 + r) * C_ + nBase + c4];
      tb[s][1] = *(const f4v*)&W[(size_t)(k0 + r + 1) * C_ + nBase + c4];
    }
#pragma unroll
    for (int s = 0; s < 2; ++s) {
      const int sb = tid + 128 * s;
      const int r = (sb >> 4) * 2, c4 = (sb & 15) * 4;
      const h4v h0 = __builtin_convertvector(tb[s][0], h4v);
      const h4v h1 = __builtin_convertvector(tb[s][1], h4v);
#pragma unroll
      for (int j = 0; j < 4; ++j) {
        h2v p; p.x = h0[j]; p.y = h1[j];
        *(h2v*)&lds_bt[(c4 + j) * 32 + r] = p;
      }
    }
    if (k0 + 32 < C_)
      __builtin_prefetch(&W[(size_t)(k0 + 32 + (tid & 31)) * C_ + nBase], 0, 0);
    __syncthreads();

    const v16h a0 = load_a_frag(&lds_a[(wave * 32 + (lane & 15)) * 32], lane);
    const v16h a1 = load_a_frag(&lds_a[(wave * 32 + 16 + (lane & 15)) * 32], lane);
#pragma unroll
    for (int nt = 0; nt < 4; ++nt) {
      const v16h b = load_b_frag(&lds_bt[(nt * 16 + (lane & 15)) * 32], lane);
      acc[0][nt] = wmma_f16(a0, b, acc[0][nt]);
      acc[1][nt] = wmma_f16(a1, b, acc[1][nt]);
    }
    __syncthreads();
  }

  const int hi = (lane & 16) ? 8 : 0;
  const int n0 = lane & 15;
#pragma unroll
  for (int mt = 0; mt < 2; ++mt)
#pragma unroll
    for (int nt = 0; nt < 4; ++nt)
#pragma unroll
      for (int r = 0; r < 8; ++r) {
        const int m = mBase + wave * 32 + mt * 16 + hi + r;
        out[(size_t)m * C_ + nBase + nt * 16 + n0] = acc[mt][nt][r];
      }
}

// ---------------------------------------------------------------------------
extern "C" void kernel_launch(void* const* d_in, const int* in_sizes, int n_in,
                              void* d_out, int out_size, void* d_ws, size_t ws_size,
                              hipStream_t stream) {
  const float* x  = (const float*)d_in[0];
  const float* wq = (const float*)d_in[1];
  const float* wk = (const float*)d_in[2];
  const float* wv = (const float*)d_in[3];
  const float* wo = (const float*)d_in[4];

  // Workspace: q16 32MB | k16 8MB | v16 8MB | attn16 32MB  (= 80MB total)
  char* ws = (char*)d_ws;
  _Float16* q16 = (_Float16*)(ws);
  _Float16* k16 = (_Float16*)(ws + (size_t)32 * 1024 * 1024);
  _Float16* v16 = (_Float16*)(ws + (size_t)40 * 1024 * 1024);
  _Float16* a16 = (_Float16*)(ws + (size_t)48 * 1024 * 1024);

  // 1) QKV projections (f16 WMMA, f32 accum), scattered to [B,H,T,HD]
  qkv_gemm_kernel<<<dim3(BT_ / 128, (NH_ * HD_) / 64), dim3(128), 0, stream>>>(
      x, wq, q16, NH_ * HD_, NH_);
  qkv_gemm_kernel<<<dim3(BT_ / 128, (NKV_ * HD_) / 64), dim3(128), 0, stream>>>(
      x, wk, k16, NKV_ * HD_, NKV_);
  qkv_gemm_kernel<<<dim3(BT_ / 128, (NKV_ * HD_) / 64), dim3(128), 0, stream>>>(
      x, wv, v16, NKV_ * HD_, NKV_);

  // 2) RoPE in place on q16 and k16
  {
    const size_t qpairs = (size_t)B_ * NH_ * T_ * 64;
    const size_t kpairs = (size_t)B_ * NKV_ * T_ * 64;
    rope_kernel<<<(unsigned)(qpairs / 256), 256, 0, stream>>>(q16, NH_);
    rope_kernel<<<(unsigned)(kpairs / 256), 256, 0, stream>>>(k16, NKV_);
  }

  // 3) Causal GQA flash attention: 4 waves / 64 queries per block share K/V
  attn_kernel<<<dim3(T_ / 64, NH_, B_), dim3(128), 0, stream>>>(q16, k16, v16, a16);

  // 4) Output projection -> f32 d_out [B*T, C]
  out_gemm_kernel<<<dim3(BT_ / 128, C_ / 64), dim3(128), 0, stream>>>(
      a16, wo, (float*)d_out);
}
